// TrueQuantizedConv2d_49297634623714
// MI455X (gfx1250) — matrix operand order repaired and checked by reading, not compiled, not run on hardware
//
#include <hip/hip_runtime.h>

typedef __attribute__((ext_vector_type(16))) _Float16 v16h;
typedef __attribute__((ext_vector_type(8)))  _Float16 v8h;
typedef __attribute__((ext_vector_type(8)))  float    v8f;
typedef __attribute__((ext_vector_type(4)))  float    f32x4;

#define BB    16
#define CIN   128
#define COUT  128
#define HH    128
#define WW    128
#define RANK  16
#define PH    130          // padded H (1-halo each side)
#define PW    130          // padded W

// async 16B global -> LDS copy (CDNA5 VGLOBAL async op, tracked by ASYNCcnt)
#define ASYNC_CP16(ldsoff, gptr)                                   \
  asm volatile("global_load_async_to_lds_b128 %0, %1, off"         \
               :: "v"(ldsoff), "v"(gptr) : "memory")
#define WAIT_ASYNC0() asm volatile("s_wait_asynccnt 0x0" ::: "memory")

// -------------------------------------------------------------------------
// Kernel 0: zero the padded NHWC f16 staging tensor (halo must be zero).
// 34,611,200 f16 = 4,326,400 v8h chunks = 16,900 blocks x 256 threads.
// -------------------------------------------------------------------------
__global__ __launch_bounds__(256) void zero_xh(_Float16* __restrict__ xh) {
  const size_t t = (size_t)blockIdx.x * 256 + threadIdx.x;
  *(v8h*)(xh + t * 8) = (v8h)(_Float16)0.0f;
}

// -------------------------------------------------------------------------
// Kernel 1: NCHW f32 -> padded NHWC f16 transpose/convert.
// Block tile: 32 cin x 64 w of one (b,h) row; LDS-tiled (stride 40 f16 = 80B,
// 16B-aligned chunks, 20-bank row skew). Coalesced float4 reads, v8h writes.
// -------------------------------------------------------------------------
__global__ __launch_bounds__(256) void nchw_to_nhwc_f16(
    const float* __restrict__ x, _Float16* __restrict__ xh) {
  __shared__ _Float16 tileT[64 * 40];              // [w][c] f16
  const int t  = threadIdx.x;
  const int bid = blockIdx.x;                      // 16*128*2*4 = 16384
  const int c0 = (bid & 3) * 32;
  const int w0 = ((bid >> 2) & 1) * 64;
  const int h  = (bid >> 3) & (HH - 1);
  const int b  = bid >> 10;

  // read: thread -> (c_local, 8 consecutive w)
  {
    const int cl = t >> 3;                         // 0..31
    const int w8 = (t & 7) * 8;
    const float* src = x + (((size_t)(b * CIN + c0 + cl) * HH + h) * WW + w0 + w8);
    const f32x4 a = *(const f32x4*)(src);
    const f32x4 c = *(const f32x4*)(src + 4);
#pragma unroll
    for (int i = 0; i < 4; ++i) {
      tileT[(w8 + i)     * 40 + cl] = (_Float16)a[i];
      tileT[(w8 + 4 + i) * 40 + cl] = (_Float16)c[i];
    }
  }
  __syncthreads();
  // write: thread -> (w_local, 8 consecutive c), 16B vector stores
  {
    const int wl = t >> 2;                         // 0..63
    const int q  = (t & 3) * 8;
    const v8h v = *(const v8h*)&tileT[wl * 40 + q];
    _Float16* dst = xh + (((size_t)(b * PH + h + 1) * PW + (w0 + wl + 1)) * CIN + c0 + q);
    *(v8h*)dst = v;
  }
}

// -------------------------------------------------------------------------
// Kernel 2: dequantize 4-bit weights + fold LoRA (B@A), store f16 weights
// pre-swizzled into the CDNA5 WMMA A-matrix (16x32 f16) lane layout:
// wswz[kc][mt][lane][j]; lane<16 holds K {0..7,16..23}, lane>=16 {8..15,24..31}.
// -------------------------------------------------------------------------
__global__ __launch_bounds__(256) void dequant_fuse_swizzle(
    const int*   __restrict__ qw, const float* __restrict__ scales,
    const float* __restrict__ lA, const float* __restrict__ lB,
    _Float16*    __restrict__ wswz) {
  const int t    = blockIdx.x * 256 + threadIdx.x;   // 0..147455
  const int j    =  t        & 15;
  const int lane = (t >> 4)  & 31;
  const int mt   = (t >> 9)  & 7;
  const int kc   =  t >> 12;                         // 0..35

  const int v = j >> 1, p = j & 1;
  const int klocal = (v < 4 ? 2 * v + p : 16 + 2 * (v - 4) + p) + ((lane & 16) ? 8 : 0);
  const int cout = mt * 16 + (lane & 15);
  const int tap  = kc >> 2;
  const int cg   = kc & 3;
  const int cin  = cg * 32 + klocal;

  const int f   = (cout * CIN + cin) * 9 + tap;
  const int q   = qw[f >> 1];
  const int nib = (f & 1) ? ((q >> 4) & 15) : (q & 15);
  float w = (-1.0f + (float)nib * (2.0f / 15.0f)) * scales[f >> 6];

  float lacc = 0.0f;
#pragma unroll
  for (int r = 0; r < RANK; ++r)
    lacc += lA[(r * CIN + cin) * 9 + tap] * lB[cout * RANK + r];

  wswz[t] = (_Float16)(w + lacc);
}

// -------------------------------------------------------------------------
// Kernel 3: implicit-GEMM 3x3 conv via v_wmma_f32_16x16x32_f16 with async
// global->LDS staging. Block = 8 waves, tile = 128 cout x 64 w of one (b,h).
// Stage (kh, cin-group): 66-row x 32-cin halo tile copied by async b128s
// (no VALU, no bounds checks -- padded NHWC f16 source), reused by 3 kw taps.
// LDS rows skewed 80B; B-fragments = two 16B ds_load_b128 per lane.
// -------------------------------------------------------------------------
#define XROW 40            // f16 per LDS row (80B = 5x16B chunks, 20-bank skew)

__global__ __launch_bounds__(256) void qconv_wmma(
    const _Float16* __restrict__ xh,    // padded NHWC f16 [B][130][130][128]
    const _Float16* __restrict__ wswz,  // swizzled f16 weights
    const float*    __restrict__ bias,
    float*          __restrict__ out) {
  __shared__ _Float16 xtile[66 * XROW];

  const int tid  = threadIdx.x;
  const int lane = tid & 31;
  const int wv   = tid >> 5;                 // wave id == M subtile
  const int bid  = blockIdx.x;               // 0..4095
  const int w0   = (bid & 1) * 64;
  const int h    = (bid >> 1) & (HH - 1);
  const int b    = bid >> 8;

  const int lane15 = lane & 15;
  const int k0     = (lane & 16) ? 8 : 0;
  const int mhi    = (lane & 16) ? 8 : 0;

  // async chunk coords: chunk c -> row n = c>>2 (halo col), 16B quarter q = c&3
  const int n1 = tid >> 2, q1 = tid & 3;              // chunks 0..255 (rows 0..63)
  const int n2 = 64 + (tid >> 2), q2 = tid & 3;       // chunks 256..263 (tid<8)
  const unsigned lds1 = (unsigned)(size_t)&xtile[n1 * XROW + q1 * 8];
  const unsigned lds2 = (unsigned)(size_t)&xtile[n2 * XROW + q2 * 8];

  v8f acc[4] = {};

  for (int kh = 0; kh < 3; ++kh) {
    const size_t rowbase = ((size_t)(b * PH + h + kh)) * PW;   // padded, no checks
    for (int cg = 0; cg < 4; ++cg) {
      __syncthreads();   // previous tile fully consumed by all waves
      {
        const _Float16* g1 = xh + ((rowbase + (w0 + n1)) * CIN + cg * 32 + q1 * 8);
        ASYNC_CP16(lds1, g1);
        if (tid < 8) {
          const _Float16* g2 = xh + ((rowbase + (w0 + n2)) * CIN + cg * 32 + q2 * 8);
          ASYNC_CP16(lds2, g2);
        }
      }
      WAIT_ASYNC0();       // this wave's copies landed in LDS
      __syncthreads();     // all waves' copies visible

#pragma unroll
      for (int kw = 0; kw < 3; ++kw) {
        const int kc = (kh * 3 + kw) * 4 + cg;
        const _Float16* wp = wswz + ((size_t)(kc * 8 + wv) * 32 + lane) * 16;
        if (kc + 1 < 36)   // keep next weight chunk warm (global_prefetch_b8)
          __builtin_prefetch(wp + 8 * 32 * 16, 0, 1);
        const v16h afrag = *(const v16h*)wp;
#pragma unroll
        for (int ns = 0; ns < 4; ++ns) {
          const int row = ns * 16 + lane15 + kw;       // 0..65
          const v8h lo = *(const v8h*)&xtile[row * XROW + k0];
          const v8h hi = *(const v8h*)&xtile[row * XROW + k0 + 16];
          const v16h bfrag = __builtin_shufflevector(
              lo, hi, 0, 1, 2, 3, 4, 5, 6, 7, 8, 9, 10, 11, 12, 13, 14, 15);
          acc[ns] = __builtin_amdgcn_wmma_f32_16x16x32_f16(
              false, afrag, false, bfrag, (short)0, acc[ns], false, false);
        }
      }
    }
  }

  // epilogue: bias + f32 store, per ISA C/D layout (VGPR i -> M = i + mhi)
#pragma unroll
  for (int ns = 0; ns < 4; ++ns) {
#pragma unroll
    for (int i = 0; i < 8; ++i) {
      const int cout = wv * 16 + i + mhi;
      const int w    = w0 + ns * 16 + lane15;
      out[(((size_t)b * COUT + cout) * HH + h) * WW + w] = acc[ns][i] + bias[cout];
    }
  }
}

// -------------------------------------------------------------------------
extern "C" void kernel_launch(void* const* d_in, const int* in_sizes, int n_in,
                              void* d_out, int out_size, void* d_ws, size_t ws_size,
                              hipStream_t stream) {
  (void)in_sizes; (void)n_in; (void)out_size; (void)ws_size;
  const float* x      = (const float*)d_in[0];
  const int*   qw     = (const int*)  d_in[1];
  const float* scales = (const float*)d_in[2];
  const float* lA     = (const float*)d_in[3];
  const float* lB     = (const float*)d_in[4];
  const float* bias   = (const float*)d_in[5];

  _Float16* wswz = (_Float16*)d_ws;                       // 288 KB
  _Float16* xh   = (_Float16*)((char*)d_ws + 147456 * 2); // 16*130*130*128 f16 (~66 MB)

  zero_xh<<<16900, 256, 0, stream>>>(xh);                 // halo (and body) zeroed
  nchw_to_nhwc_f16<<<16384, 256, 0, stream>>>(x, xh);     // interior convert/transpose
  dequant_fuse_swizzle<<<576, 256, 0, stream>>>(qw, scales, lA, lB, wswz);
  qconv_wmma<<<4096, 256, 0, stream>>>(xh, wswz, bias, (float*)d_out);
}
